// Update_54631984005677
// MI455X (gfx1250) — matrix operand order repaired and checked by reading, not compiled, not run on hardware
//
#include <hip/hip_runtime.h>

// ---------------------------------------------------------------------------
// Problem constants
// ---------------------------------------------------------------------------
constexpr int BATCH = 4;
constexpr int H     = 256;
constexpr int W     = 320;
constexpr int HW    = H * W;          // 81920
constexpr int HD    = 32;
constexpr int NS    = 256;
constexpr int K_GRU  = 65 * 9;        // 585
constexpr int K_GRUP = 608;           // padded to 19*32
constexpr int NC_GRU = 19;
constexpr int K_DIL  = 32 * 9;        // 288 = 9*32 exactly
constexpr int NC_DIL = 9;

// d_out layout (float32: h, nd, prob, conf, conf0 concatenated)
constexpr size_t N_H       = (size_t)BATCH * HD * HW;        // 10,485,760
constexpr size_t OFF_ND    = N_H;
constexpr size_t OFF_PROB  = OFF_ND + (size_t)BATCH * HW;    // 10,813,440
constexpr size_t OFF_CONF  = OFF_PROB + (size_t)BATCH * NS * HW;
constexpr size_t OFF_CONF0 = OFF_CONF + (size_t)BATCH * HW;

// Scratch carved out of the prob region (fully rewritten by heads_kernel):
// zstash (f32, N_H floats), then f16 area: xpk(65ch), rhpk, hpk, packed weights
constexpr size_t XPK_OFF  = 0;                                  // halves
constexpr size_t XPK_N    = (size_t)BATCH * 65 * HW;            // 21,299,200
constexpr size_t RHPK_OFF = XPK_OFF + XPK_N;
constexpr size_t PLANE_N  = (size_t)BATCH * 32 * HW;            // 10,485,760
constexpr size_t HPK_OFF  = RHPK_OFF + PLANE_N;
constexpr size_t WPK_OFF  = HPK_OFF + PLANE_N;
// packed weight sub-offsets (halves)
constexpr int WO_ZR = 0;                  // [64][608]
constexpr int WO_Q  = WO_ZR + 64 * K_GRUP;// [32][608]
constexpr int WO_DC = WO_Q + 32 * K_GRUP; // [64][288]
constexpr int WO_2  = WO_DC + 64 * K_DIL; // [64][32]
constexpr int WO_3  = WO_2 + 64 * 32;     // [256][64]
constexpr int W_TOT = WO_3 + 256 * 64;    // 95,232

typedef _Float16 f16;
typedef _Float16 v8h  __attribute__((ext_vector_type(8)));
typedef _Float16 v16h __attribute__((ext_vector_type(16)));
typedef float    v8f  __attribute__((ext_vector_type(8)));
typedef unsigned int u32x4 __attribute__((ext_vector_type(4)));
typedef int      i32x4 __attribute__((ext_vector_type(4)));
typedef int      i32x8 __attribute__((ext_vector_type(8)));

__device__ __forceinline__ int imin(int a, int b) { return a < b ? a : b; }
__device__ __forceinline__ int imax(int a, int b) { return a > b ? a : b; }

// ---------------------------------------------------------------------------
// Tensor Data Mover: 2D weight tile (rows x 32 f16) global -> LDS, with a 16B
// row pad giving the 40-half LDS row stride the fragments expect.
// ---------------------------------------------------------------------------
__device__ __forceinline__ void tdm_load_w_tile(const f16* gsrc, void* lds_dst,
                                                int rows, int row_stride_elems) {
    unsigned long long ga = (unsigned long long)(size_t)gsrc;
    unsigned int lds = (unsigned int)(size_t)lds_dst;
    u32x4 g0;
    g0[0] = 1u;                                            // count=1 (user D#)
    g0[1] = lds;                                           // lds_addr
    g0[2] = (unsigned int)ga;                              // global_addr[31:0]
    g0[3] = (unsigned int)((ga >> 32) & 0x01FFFFFFu) | (2u << 30);  // [56:32] | type=2
    i32x8 g1;
    // data_size=1(2B)<<16 | pad_enable<<20 | pad_interval=3(64B)<<22 | pad_amount=3(16B)<<25
    g1[0] = (1 << 16) | (1 << 20) | (3 << 22) | (3 << 25);
    g1[1] = (32 & 0xFFFF) << 16;                           // tensor_dim0 = 32
    g1[2] = (rows & 0xFFFF) << 16;                         // tensor_dim1 = rows
    g1[3] = (32 << 16);                                    // tile_dim0 = 32
    g1[4] = rows & 0xFFFF;                                 // tile_dim1 = rows
    g1[5] = row_stride_elems;                              // tensor_dim0_stride
    g1[6] = 0;
    g1[7] = 0;
    i32x4 g2 = {0, 0, 0, 0};
    i32x4 g3 = {0, 0, 0, 0};
#if __clang_major__ >= 23
    i32x8 g4 = {0, 0, 0, 0, 0, 0, 0, 0};
    __builtin_amdgcn_tensor_load_to_lds(g0, g1, g2, g3, g4, 0);
#else
    __builtin_amdgcn_tensor_load_to_lds(g0, g1, g2, g3, 0);
#endif
}

// ---------------------------------------------------------------------------
// WMMA fragment loads from LDS (wave32 layouts per CDNA5 ISA 7.12.2)
// ---------------------------------------------------------------------------
__device__ __forceinline__ v16h load_a_frag(const f16* As, int stride, int mtile, int lane) {
    const f16* row = As + (mtile * 16 + (lane & 15)) * stride;
    int kh = (lane >> 4) * 8;
    v8h lo = *(const v8h*)(row + kh);
    v8h hi = *(const v8h*)(row + 16 + kh);
    return __builtin_shufflevector(lo, hi, 0,1,2,3,4,5,6,7,8,9,10,11,12,13,14,15);
}
__device__ __forceinline__ v16h load_b_frag(const f16* Bs, int stride, int ntile, int lane, int koff) {
    const f16* row = Bs + (ntile * 16 + (lane & 15)) * stride + koff + (lane >> 4) * 8;
    v8h lo = *(const v8h*)(row);
    v8h hi = *(const v8h*)(row + 16);
    return __builtin_shufflevector(lo, hi, 0,1,2,3,4,5,6,7,8,9,10,11,12,13,14,15);
}

__device__ __forceinline__ float sigmoidf_fast(float x) { return 1.f / (1.f + __expf(-x)); }
__device__ __forceinline__ float tanhf_fast(float x) {
    float xc = fminf(fmaxf(x, -15.f), 15.f);
    float e = __expf(2.f * xc);
    return (e - 1.f) / (e + 1.f);
}

// GRU patch staging: (c,tap) math forced to SALU via readfirstlane (k is
// wave-uniform: k = k0 + i*8 + wv); only the x clamp/mask stays per-lane.
__device__ __forceinline__ void stage_gru_patch(f16* Bs, const f16* base_lo, const f16* base_hi,
                                                int y, int x0, int lane, int wv, int k0) {
    #pragma unroll
    for (int i = 0; i < 4; ++i) {
        int kk = i * 8 + wv;
        int k = __builtin_amdgcn_readfirstlane(k0 + kk);
        int kcl = imin(k, K_GRU - 1);
        int c = kcl / 9, tap = kcl - c * 9;
        int dy = tap / 3 - 1, dx = (tap % 3) - 1;
        int yy = y + dy;
        int yc = imin(imax(yy, 0), H - 1);
        const f16* rowp = (c < 32 ? base_lo : base_hi) + (size_t)c * HW + yc * W;
        bool vrow = ((unsigned)yy < (unsigned)H) && (k < K_GRU);
        #pragma unroll
        for (int hh = 0; hh < 2; ++hh) {
            int p = lane + hh * 32;
            int xx = x0 + p + dx;
            int xc = imin(imax(xx, 0), W - 1);
            f16 t = rowp[xc];
            bool ok = vrow && ((unsigned)xx < (unsigned)W);
            Bs[p * 40 + kk] = ok ? t : (f16)0;
        }
    }
}

// ---------------------------------------------------------------------------
// Pack kernels
// ---------------------------------------------------------------------------
__global__ __launch_bounds__(256) void pack_x_kernel(
    const float* __restrict__ hidden, const float* __restrict__ ndep,
    const float* __restrict__ corr, f16* __restrict__ xpk)
{
    size_t i = (size_t)blockIdx.x * 256 + threadIdx.x;
    if (i >= XPK_N) return;
    int pix = (int)(i % HW);
    size_t t = i / HW;
    int c = (int)(t % 65), b = (int)(t / 65);
    float v;
    if (c < 32)       v = hidden[((size_t)b * 32 + c) * HW + pix];
    else if (c == 32) v = ndep[(size_t)b * HW + pix];
    else              v = corr[((size_t)b * 32 + (c - 33)) * HW + pix];
    xpk[i] = (f16)v;
}

__global__ __launch_bounds__(256) void pack_w_kernel(
    const float* __restrict__ wz, const float* __restrict__ wr,
    const float* __restrict__ wq, const float* __restrict__ d1,
    const float* __restrict__ c1, const float* __restrict__ d2,
    const float* __restrict__ d3, f16* __restrict__ wpk)
{
    int i = blockIdx.x * 256 + threadIdx.x;
    if (i >= W_TOT) return;
    float v;
    if (i < WO_Q) {
        int o = i / K_GRUP, k = i - o * K_GRUP;
        v = (k < K_GRU) ? (o < 32 ? wz[o * K_GRU + k] : wr[(o - 32) * K_GRU + k]) : 0.f;
    } else if (i < WO_DC) {
        int j = i - WO_Q;
        int o = j / K_GRUP, k = j - o * K_GRUP;
        v = (k < K_GRU) ? wq[o * K_GRU + k] : 0.f;
    } else if (i < WO_2) {
        int j = i - WO_DC;
        int o = j / K_DIL, k = j - o * K_DIL;
        v = (o < 32) ? d1[o * K_DIL + k] : c1[(o - 32) * K_DIL + k];
    } else if (i < WO_3) {
        v = d2[i - WO_2];
    } else {
        v = d3[i - WO_3];
    }
    wpk[i] = (f16)v;
}

// ---------------------------------------------------------------------------
// Kernel 1: z = sigmoid(convz), rh = sigmoid(convr) * hidden
// ---------------------------------------------------------------------------
__global__ __launch_bounds__(256) void gru_zr_kernel(
    const f16* __restrict__ xpk, const f16* __restrict__ wzr,
    const float* __restrict__ bz, const float* __restrict__ br,
    float* __restrict__ zstash, f16* __restrict__ rhpk)
{
    __shared__ f16 As[64 * 40];
    __shared__ f16 Bs[64 * 40];

    int gx = blockIdx.x;
    int tx = gx % 5, y = (gx / 5) & (H - 1), b = gx / (5 * H);
    int x0 = tx * 64;
    int tid = threadIdx.x, lane = tid & 31, wv = tid >> 5;
    int mtile = wv >> 1, nt0 = (wv & 1) * 2;
    const f16* xb = xpk + (size_t)b * 65 * HW;

    v8f acc0 = {}, acc1 = {};
    for (int kc = 0; kc < NC_GRU; ++kc) {
        int k0 = kc * 32;
        if (wv == 0) tdm_load_w_tile(wzr + k0, As, 64, K_GRUP);
        stage_gru_patch(Bs, xb, xb, y, x0, lane, wv, k0);
        if (wv == 0) __builtin_amdgcn_s_wait_tensorcnt(0);
        __syncthreads();
        v16h af = load_a_frag(As, 40, mtile, lane);
        v16h b0 = load_b_frag(Bs, 40, nt0, lane, 0);
        v16h b1 = load_b_frag(Bs, 40, nt0 + 1, lane, 0);
        acc0 = __builtin_amdgcn_wmma_f32_16x16x32_f16(false, af, false, b0, (short)0, acc0, false, false);
        acc1 = __builtin_amdgcn_wmma_f32_16x16x32_f16(false, af, false, b1, (short)0, acc1, false, false);
        __syncthreads();
    }

    int mb = mtile * 16 + ((lane >> 4) ? 8 : 0);
    #pragma unroll
    for (int nt = 0; nt < 2; ++nt) {
        int x = x0 + (nt0 + nt) * 16 + (lane & 15);
        const v8f& acc = nt ? acc1 : acc0;
        #pragma unroll
        for (int v = 0; v < 8; ++v) {
            int o = mb + v;
            size_t pi = (((size_t)b * 32 + (o & 31)) * H + y) * W + x;
            if (o < 32) {
                zstash[pi] = sigmoidf_fast(acc[v] + bz[o]);
            } else {
                float r = sigmoidf_fast(acc[v] + br[o - 32]);
                rhpk[pi] = (f16)(r * (float)xb[(size_t)(o - 32) * HW + y * W + x]);
            }
        }
    }
}

// ---------------------------------------------------------------------------
// Kernel 2: q = tanh(convq([rh, x])); h = (1-z)*hidden + z*q
// ---------------------------------------------------------------------------
__global__ __launch_bounds__(256) void gru_qh_kernel(
    const f16* __restrict__ xpk, const f16* __restrict__ rhpk,
    const f16* __restrict__ wq, const float* __restrict__ bq,
    const float* __restrict__ hidden, const float* __restrict__ zstash,
    float* __restrict__ out_h, f16* __restrict__ hpk)
{
    __shared__ f16 As[32 * 40];
    __shared__ f16 Bs[64 * 40];

    int gx = blockIdx.x;
    int tx = gx % 5, y = (gx / 5) & (H - 1), b = gx / (5 * H);
    int x0 = tx * 64;
    int tid = threadIdx.x, lane = tid & 31, wv = tid >> 5;
    int mtile = wv >> 2, ntile = wv & 3;
    const f16* xb = xpk + (size_t)b * 65 * HW;
    const f16* rb = rhpk + (size_t)b * 32 * HW;

    v8f acc = {};
    for (int kc = 0; kc < NC_GRU; ++kc) {
        int k0 = kc * 32;
        if (wv == 0) tdm_load_w_tile(wq + k0, As, 32, K_GRUP);
        stage_gru_patch(Bs, rb, xb, y, x0, lane, wv, k0);
        if (wv == 0) __builtin_amdgcn_s_wait_tensorcnt(0);
        __syncthreads();
        v16h af = load_a_frag(As, 40, mtile, lane);
        v16h bf = load_b_frag(Bs, 40, ntile, lane, 0);
        acc = __builtin_amdgcn_wmma_f32_16x16x32_f16(false, af, false, bf, (short)0, acc, false, false);
        __syncthreads();
    }

    int x = x0 + ntile * 16 + (lane & 15);
    int mb = mtile * 16 + ((lane >> 4) ? 8 : 0);
    #pragma unroll
    for (int v = 0; v < 8; ++v) {
        int o = mb + v;
        size_t pi = (((size_t)b * 32 + o) * H + y) * W + x;
        float q = tanhf_fast(acc[v] + bq[o]);
        float z = zstash[pi];
        float h = (1.f - z) * hidden[pi] + z * q;
        out_h[pi] = h;
        hpk[pi] = (f16)h;
    }
}

// ---------------------------------------------------------------------------
// Kernel 3: heads. WMMA phases + lane-parallel softmax/regression epilogue.
// ---------------------------------------------------------------------------
__global__ __launch_bounds__(256) void heads_kernel(
    const f16* __restrict__ hpk, const f16* __restrict__ wpk,
    const float* __restrict__ dh3_b, const float* __restrict__ ch2_w,
    const float* __restrict__ ch2_b, float* __restrict__ out)
{
    __shared__ f16   AsA[64 * 40];
    __shared__ f16   BsA[32 * 40];
    __shared__ f16   t1s[32 * 40];
    __shared__ f16   As2[64 * 40];
    __shared__ f16   t2s[32 * 72];
    __shared__ f16   AsC[256 * 40];
    __shared__ float c1s[32 * 32];
    __shared__ f16   lgs[32 * 266];
    __shared__ float sm_inv[32];

    int gx = blockIdx.x;
    int tx = gx % 10, y = (gx / 10) & (H - 1), b = gx / (10 * H);
    int x0 = tx * 32;
    int tid = threadIdx.x, lane = tid & 31, wv = tid >> 5;
    int mtile = wv >> 1, ntile = wv & 1;
    int nloc = ntile * 16 + (lane & 15);
    int mb   = mtile * 16 + ((lane >> 4) ? 8 : 0);
    const f16* hb = hpk + (size_t)b * 32 * HW;

    // prefetch dh2 weights (buffer unused during phase A)
    if (wv == 0) tdm_load_w_tile(wpk + WO_2, As2, 64, 32);

    // ---- Phase A: C[64 x 32] = [dh1;ch1] x patch(h, dil=2), K = 288 ----
    v8f acc1 = {};
    for (int kc = 0; kc < NC_DIL; ++kc) {
        int k0 = kc * 32;
        if (wv == 1) tdm_load_w_tile(wpk + WO_DC + k0, AsA, 64, K_DIL);
        #pragma unroll
        for (int i = 0; i < 4; ++i) {             // scalarized dil-2 gather
            int kk = i * 8 + wv;
            int k = __builtin_amdgcn_readfirstlane(k0 + kk);
            int c = k / 9, tap = k - c * 9;
            int dy = (tap / 3 - 1) * 2, dx = (tap % 3 - 1) * 2;
            int yy = y + dy;
            int yc = imin(imax(yy, 0), H - 1);
            const f16* rowp = hb + (size_t)c * HW + yc * W;
            bool vrow = (unsigned)yy < (unsigned)H;
            int xx = x0 + lane + dx;
            int xc = imin(imax(xx, 0), W - 1);
            f16 t = rowp[xc];
            bool ok = vrow && ((unsigned)xx < (unsigned)W);
            BsA[lane * 40 + kk] = ok ? t : (f16)0;
        }
        if (wv == 1) __builtin_amdgcn_s_wait_tensorcnt(0);
        __syncthreads();
        v16h af = load_a_frag(AsA, 40, mtile, lane);
        v16h bf = load_b_frag(BsA, 40, ntile, lane, 0);
        acc1 = __builtin_amdgcn_wmma_f32_16x16x32_f16(false, af, false, bf, (short)0, acc1, false, false);
        __syncthreads();
    }
    #pragma unroll
    for (int v = 0; v < 8; ++v) {
        int o = mb + v;
        float r = fmaxf(acc1[v], 0.f);
        if (o < 32) t1s[nloc * 40 + o] = (f16)r;
        else        c1s[nloc * 32 + (o - 32)] = r;
    }
    if (wv == 0) __builtin_amdgcn_s_wait_tensorcnt(0);   // As2 ready
    __syncthreads();

    // ---- Phase B: t2[64 x 32] = dh2_w x t1, relu ----
    {
        v8f acc2 = {};
        v16h af = load_a_frag(As2, 40, mtile, lane);
        v16h bf = load_b_frag(t1s, 40, ntile, lane, 0);
        acc2 = __builtin_amdgcn_wmma_f32_16x16x32_f16(false, af, false, bf, (short)0, acc2, false, false);
        #pragma unroll
        for (int v = 0; v < 8; ++v)
            t2s[nloc * 72 + mb + v] = (f16)fmaxf(acc2[v], 0.f);
    }
    __syncthreads();

    // ---- Phase C: logits[256 x 32] = dh3_w[256 x 64] x t2 + b ----
    v8f acc3[4] = {{}, {}, {}, {}};
    for (int kc = 0; kc < 2; ++kc) {
        if (wv == 0) {
            tdm_load_w_tile(wpk + WO_3 + kc * 32, AsC, 256, 64);
            __builtin_amdgcn_s_wait_tensorcnt(0);
        }
        __syncthreads();
        #pragma unroll
        for (int mi = 0; mi < 2; ++mi) {
            v16h af = load_a_frag(AsC, 40, wv * 2 + mi, lane);
            #pragma unroll
            for (int nt = 0; nt < 2; ++nt) {
                v16h bf = load_b_frag(t2s, 72, nt, lane, kc * 32);
                acc3[mi * 2 + nt] = __builtin_amdgcn_wmma_f32_16x16x32_f16(
                    false, af, false, bf, (short)0, acc3[mi * 2 + nt], false, false);
            }
        }
        __syncthreads();
    }
    #pragma unroll
    for (int mi = 0; mi < 2; ++mi)
        #pragma unroll
        for (int nt = 0; nt < 2; ++nt) {
            int chb = (wv * 2 + mi) * 16 + ((lane >> 4) ? 8 : 0);
            int p = nt * 16 + (lane & 15);
            #pragma unroll
            for (int v = 0; v < 8; ++v) {
                int ch = chb + v;
                lgs[p * 266 + ch] = (f16)(acc3[mi * 2 + nt][v] + dh3_b[ch]);
            }
        }
    __syncthreads();

    // ---- Lane-parallel softmax / window regression / confidence ----
    {
        int lg = lane & 7;                   // lane-in-group (8 lanes per pixel)
        int p  = wv * 4 + (lane >> 3);       // pixel 0..31
        f16* row = lgs + p * 266;

        float m = -3.0e38f; int am = 0;      // max/argmax (32 channels per lane)
        #pragma unroll 8
        for (int j = 0; j < 32; ++j) {
            int ch = lg * 32 + j;
            float v = (float)row[ch];
            if (v > m) { m = v; am = ch; }
        }
        #pragma unroll
        for (int off = 4; off >= 1; off >>= 1) {
            float mo = __shfl_xor(m, off, 32);
            int   ao = __shfl_xor(am, off, 32);
            if (mo > m || (mo == m && ao < am)) { m = mo; am = ao; }
        }
        float s = 0.f;                       // exp + sum; keep exp in LDS
        #pragma unroll 8
        for (int j = 0; j < 32; ++j) {
            int ch = lg * 32 + j;
            float e = __expf((float)row[ch] - m);
            s += e;
            row[ch] = (f16)e;
        }
        #pragma unroll
        for (int off = 4; off >= 1; off >>= 1) s += __shfl_xor(s, off, 32);
        float inv = 1.f / s;

        float cs = 0.f;                      // confidence dot, 4 ch per lane
        #pragma unroll
        for (int t = 0; t < 4; ++t) {
            int c = lg * 4 + t;
            cs += c1s[p * 32 + c] * ch2_w[c];
        }
        #pragma unroll
        for (int off = 4; off >= 1; off >>= 1) cs += __shfl_xor(cs, off, 32);

        if (lg == 0) {
            int x = x0 + p;
            size_t pix = ((size_t)b * H + y) * W + x;
            float c0 = cs + ch2_b[0];
            out[OFF_CONF0 + pix] = c0;
            out[OFF_CONF + pix]  = sigmoidf_fast(c0);
            float num = 0.f, den = 1e-6f;
            #pragma unroll
            for (int j = -4; j <= 4; ++j) {
                int idx = imin(imax(am + j, 0), NS - 1);
                float pj = (float)row[idx] * inv;
                num += (float)idx * pj;
                den += pj;
            }
            out[OFF_ND + pix] = (num / den) / (float)(NS - 1);
            sm_inv[p] = inv;
        }
    }
    __syncthreads();

    // ---- Coalesced prob stores: wave -> channel, lane -> pixel ----
    {
        int x = x0 + lane;
        float inv = sm_inv[lane];
        const f16* col = lgs + lane * 266;
        #pragma unroll 4
        for (int it = 0; it < 32; ++it) {
            int ch = it * 8 + wv;
            out[OFF_PROB + (((size_t)b * NS + ch) * H + y) * W + x] = (float)col[ch] * inv;
        }
    }
}

// ---------------------------------------------------------------------------
// Launch
// ---------------------------------------------------------------------------
extern "C" void kernel_launch(void* const* d_in, const int* in_sizes, int n_in,
                              void* d_out, int out_size, void* d_ws, size_t ws_size,
                              hipStream_t stream) {
    (void)in_sizes; (void)n_in; (void)out_size; (void)d_ws; (void)ws_size;
    const float* hidden = (const float*)d_in[0];
    const float* ndep   = (const float*)d_in[1];
    const float* corr   = (const float*)d_in[2];
    const float* wz     = (const float*)d_in[3];
    const float* bz     = (const float*)d_in[4];
    const float* wr     = (const float*)d_in[5];
    const float* br     = (const float*)d_in[6];
    const float* wq     = (const float*)d_in[7];
    const float* bq     = (const float*)d_in[8];
    const float* dh1w   = (const float*)d_in[9];
    const float* dh2w   = (const float*)d_in[10];
    const float* dh3w   = (const float*)d_in[11];
    const float* dh3b   = (const float*)d_in[12];
    const float* ch1w   = (const float*)d_in[13];
    const float* ch2w   = (const float*)d_in[14];
    const float* ch2b   = (const float*)d_in[15];
    float* out = (float*)d_out;

    float* zstash = out + OFF_PROB;
    f16*   fbase  = (f16*)(out + OFF_PROB + N_H);
    f16* xpk  = fbase + XPK_OFF;
    f16* rhpk = fbase + RHPK_OFF;
    f16* hpk  = fbase + HPK_OFF;
    f16* wpk  = fbase + WPK_OFF;

    const int xblocks = (int)((XPK_N + 255) / 256);
    pack_x_kernel<<<xblocks, 256, 0, stream>>>(hidden, ndep, corr, xpk);
    pack_w_kernel<<<(W_TOT + 255) / 256, 256, 0, stream>>>(wz, wr, wq, dh1w, ch1w,
                                                           dh2w, dh3w, wpk);

    const int gruBlocks  = BATCH * H * (W / 64);   // 5120
    const int headBlocks = BATCH * H * (W / 32);   // 10240
    gru_zr_kernel<<<gruBlocks, 256, 0, stream>>>(xpk, wpk + WO_ZR, bz, br, zstash, rhpk);
    gru_qh_kernel<<<gruBlocks, 256, 0, stream>>>(xpk, rhpk, wpk + WO_Q, bq,
                                                 hidden, zstash, out /*h*/, hpk);
    heads_kernel<<<headBlocks, 256, 0, stream>>>(hpk, wpk, dh3b, ch2w, ch2b, out);
}
